// Attention1d_73718818669266
// MI455X (gfx1250) — matrix operand order repaired and checked by reading, not compiled, not run on hardware
//
#include <hip/hip_runtime.h>

// Problem constants (match reference)
#define BB 4
#define QCH 256
#define KCH 256
#define OCD 128   // per-head dim
#define NH 8
#define NXP 2048
#define NYP 2048

typedef __attribute__((ext_vector_type(16))) __bf16 v16bf;
typedef __attribute__((ext_vector_type(8)))  __bf16 v8bf;
typedef __attribute__((ext_vector_type(8)))  float  v8f;
typedef int v4i_vec __attribute__((vector_size(4 * sizeof(int))));

union Frag16 { v16bf v; v8bf h[2]; };

#if defined(__gfx1250__) && __has_builtin(__builtin_amdgcn_global_load_async_to_lds_b128)
#define HAS_ASYNC 1
#else
#define HAS_ASYNC 0
#endif

__device__ __forceinline__ __bf16 f2bf(float f) {
  union { float f; unsigned u; } in; in.f = f;
  unsigned r = in.u + 0x7FFFu + ((in.u >> 16) & 1u);   // round-to-nearest-even
  union { unsigned short s; __bf16 b; } out; out.s = (unsigned short)(r >> 16);
  return out.b;
}

__device__ __forceinline__ v8f wmma_bf16(v16bf a, v16bf b, v8f c) {
  // v_wmma_f32_16x16x32_bf16: D = A(16x32) * B(32x16) + C(16x16 f32)
  return __builtin_amdgcn_wmma_f32_16x16x32_bf16(false, a, false, b, (short)0, c, false, false);
}

__device__ __forceinline__ v8f vzero8() {
  v8f z;
#pragma unroll
  for (int i = 0; i < 8; ++i) z[i] = 0.0f;
  return z;
}

#if HAS_ASYNC
// One lane moves 16 bytes global->LDS per call (whole wave: 512B / instruction).
// INST_OFFSET applies to both the global and LDS addresses (ISA 08 §4.4).
// Builtin wants (v4i addrspace(1)*, v4i addrspace(3)*, const int, const int);
// go through uintptr_t so no implicit pointer conversions are needed.
template <int OFF>
__device__ __forceinline__ void async16(const __bf16* g, __bf16* l) {
  __builtin_amdgcn_global_load_async_to_lds_b128(
      (__attribute__((address_space(1))) v4i_vec*)(uintptr_t)g,
      (__attribute__((address_space(3))) v4i_vec*)(unsigned)(uintptr_t)l,
      OFF, 0);
}
__device__ __forceinline__ void wait_async_all() {
  asm volatile("s_wait_asynccnt 0x0" ::: "memory");
}
#else
__device__ __forceinline__ void wait_async_all() {}
#endif

// ---------------------------------------------------------------------------
// Elementwise f32 -> bf16 cast (weights)
// ---------------------------------------------------------------------------
__global__ void cast_bf16_kernel(const float* __restrict__ src, __bf16* __restrict__ dst, int n) {
  int i = blockIdx.x * blockDim.x + threadIdx.x;
  if (i < n) dst[i] = f2bf(src[i]);
}

// ---------------------------------------------------------------------------
// [b][C][N] f32  ->  [b][N][C] bf16   (tiled via LDS)
// ---------------------------------------------------------------------------
__global__ void __launch_bounds__(256) transpose_cast_kernel(const float* __restrict__ src,
                                                             __bf16* __restrict__ dst,
                                                             int C, int N) {
  __shared__ float tile[32][33];
  const int b  = blockIdx.z;
  const int n0 = blockIdx.x * 32;
  const int c0 = blockIdx.y * 32;
  const int tx = threadIdx.x;   // 0..31
  const int ty = threadIdx.y;   // 0..7
  const float* s = src + (size_t)b * C * N;
#pragma unroll
  for (int r = 0; r < 32; r += 8)
    tile[ty + r][tx] = s[(size_t)(c0 + ty + r) * N + n0 + tx];
  __syncthreads();
  __bf16* d = dst + (size_t)b * N * C;
#pragma unroll
  for (int r = 0; r < 32; r += 8)
    d[(size_t)(n0 + ty + r) * C + c0 + tx] = f2bf(tile[tx][ty + r]);
}

// ---------------------------------------------------------------------------
// Generic bf16 WMMA GEMM:  D[M,N] = W[M,CK] * Xt[b][N,CK]^T
// CK is a compile-time constant -> K loop fully unrolled, kb folded into the
// 24-bit instruction offsets of precomputed per-row base addresses.
//   MODE 0: out bf16 as [b][h][n][d]  (h = m/128, d = m%128)   -> Q^T / K^T
//   MODE 1: out bf16 as [b][h][d][n]                           -> V
//   MODE 2: out f32  as [b][m][n]                              -> final
// Block: 256 threads (8 waves). Wave w computes rows [m0,m0+16) x cols [n0,n0+128).
// ---------------------------------------------------------------------------
template <int CK, int MODE>
__global__ void __launch_bounds__(256) gemm_bf16_wmma(const __bf16* __restrict__ W,
                                                      const __bf16* __restrict__ Xt,
                                                      void* __restrict__ out,
                                                      int M, int N) {
  const int lane = threadIdx.x & 31;
  const int wave = threadIdx.x >> 5;
  const int lo = lane & 15, hi = lane >> 4;
  const int b  = blockIdx.z;
  const int m0 = (blockIdx.y * 8 + wave) * 16;
  const int n0 = blockIdx.x * 128;
  const __bf16* Xb = Xt + (size_t)b * N * CK;

  // A-row base (lane holds row m0+lo), B-row bases (lane holds col n0+t*16+lo)
  const __bf16* abase = W + (size_t)(m0 + lo) * CK + hi * 16;
  const __bf16* bbase[8];
#pragma unroll
  for (int t = 0; t < 8; ++t)
    bbase[t] = Xb + (size_t)(n0 + t * 16 + lo) * CK + hi * 16;

  v8f acc[8];
#pragma unroll
  for (int t = 0; t < 8; ++t) acc[t] = vzero8();

#pragma unroll
  for (int kb = 0; kb < CK; kb += 32) {
    Frag16 a;  // A 16x32: 16 contiguous K at (hi*16)
    a.h[0] = *(const v8bf*)(abase + kb);
    a.h[1] = *(const v8bf*)(abase + kb + 8);
#pragma unroll
    for (int t = 0; t < 8; ++t) {
      v16bf bf = *(const v16bf*)(bbase[t] + kb);  // B 32x16: 16 contiguous K
      acc[t] = wmma_bf16(a.v, bf, acc[t]);
    }
  }

  if (MODE == 2) {
    float* o = (float*)out;
#pragma unroll
    for (int t = 0; t < 8; ++t)
#pragma unroll
      for (int i = 0; i < 8; ++i) {
        int m = m0 + hi * 8 + i;
        int n = n0 + t * 16 + lo;
        o[((size_t)b * M + m) * N + n] = acc[t][i];
      }
  } else if (MODE == 0) {
    __bf16* o = (__bf16*)out;
#pragma unroll
    for (int t = 0; t < 8; ++t)
#pragma unroll
      for (int i = 0; i < 8; ++i) {
        int m = m0 + hi * 8 + i;
        int h = m >> 7, d = m & 127;
        int n = n0 + t * 16 + lo;
        o[(((size_t)b * NH + h) * N + n) * OCD + d] = f2bf(acc[t][i]);
      }
  } else {
    __bf16* o = (__bf16*)out;
#pragma unroll
    for (int t = 0; t < 8; ++t)
#pragma unroll
      for (int i = 0; i < 8; ++i) {
        int m = m0 + hi * 8 + i;
        int h = m >> 7, d = m & 127;
        int n = n0 + t * 16 + lo;
        o[(((size_t)b * NH + h) * OCD + d) * (size_t)N + n] = f2bf(acc[t][i]);
      }
  }
}

// ---------------------------------------------------------------------------
// Flash attention: per (b,h), Q^T [Nx,128], K^T [Ny,128], V [128,Ny] (bf16).
// 8 waves / block share one (b,h); each wave owns a 16-row Q tile.
// K/V chunks (32x128 and 128x32) are staged in LDS once per block via
// async-to-LDS DMA (ASYNCcnt), double-buffered: DMA(c+1) overlaps compute(c).
// Output: OT [b][x][h*128+d] bf16.
// ---------------------------------------------------------------------------
__global__ void __launch_bounds__(256) attn_flash_kernel(const __bf16* __restrict__ QT,
                                                         const __bf16* __restrict__ KT,
                                                         const __bf16* __restrict__ V,
                                                         __bf16* __restrict__ OT) {
  __shared__ __bf16 kbuf[2][32][128];  // [buf][y][d]   8 KB each
  __shared__ __bf16 vbuf[2][128][32];  // [buf][d][y]   8 KB each
  __shared__ __bf16 pbuf[8][16][32];   // per-wave P transpose (C-layout -> A-layout)

  const int tid  = threadIdx.x;
  const int lane = tid & 31;
  const int wave = tid >> 5;
  const int lo = lane & 15, hi = lane >> 4;
  const int b = blockIdx.z, h = blockIdx.y;
  const int bh = b * NH + h;
  const int x0 = blockIdx.x * 128 + wave * 16;
  const float scale = 0.08838834764831845f;  // 1/sqrt(128)

  // --- staging: each of 256 threads moves 32B of K and 32B of V per chunk ---
  const int kr = tid >> 3, kc = (tid & 7) * 16;  // K chunk: 32 rows x 128 cols
  const int vr = tid >> 1, vc = (tid & 1) * 16;  // V chunk: 128 rows x 32 cols
  const __bf16* kg0 = KT + ((size_t)bh * NYP + kr) * OCD + kc;
  const __bf16* vg0 = V + ((size_t)bh * OCD + vr) * NYP + vc;

  auto stage = [&](int buf, int y0) {
    const __bf16* kg = kg0 + (size_t)y0 * OCD;
    const __bf16* vg = vg0 + y0;
    __bf16* kd = &kbuf[buf][kr][kc];
    __bf16* vd = &vbuf[buf][vr][vc];
#if HAS_ASYNC
    async16<0>(kg, kd);
    async16<16>(kg, kd);
    async16<0>(vg, vd);
    async16<16>(vg, vd);
#else
    *(v16bf*)kd = *(const v16bf*)kg;
    *(v16bf*)vd = *(const v16bf*)vg;
#endif
  };

  // Preload Q fragments (A-layout) for this wave's 16 rows: 4 chunks of K=32.
  Frag16 qf[4];
#pragma unroll
  for (int dk = 0; dk < 4; ++dk) {
    const __bf16* qrow = QT + ((size_t)bh * NXP + x0 + lo) * OCD + dk * 32 + hi * 16;
    qf[dk].h[0] = *(const v8bf*)(qrow);
    qf[dk].h[1] = *(const v8bf*)(qrow + 8);
  }

  v8f accO[8];
#pragma unroll
  for (int t = 0; t < 8; ++t) accO[t] = vzero8();
  float mrun[8], lsum[8];
#pragma unroll
  for (int i = 0; i < 8; ++i) { mrun[i] = -3.0e38f; lsum[i] = 0.0f; }

  stage(0, 0);  // prologue DMA

  const int NCH = NYP / 32;
  for (int c = 0; c < NCH; ++c) {
    const int cur = c & 1;
    wait_async_all();   // my DMAs for chunk c have landed
    __syncthreads();    // everyone's have; everyone done reading buf[1-cur]
    if (c + 1 < NCH) stage(1 - cur, (c + 1) * 32);  // overlaps compute below

    // ---- S = Q K^T for two 16-col tiles (from LDS) --------------------------
    v8f s0 = vzero8(), s1 = vzero8();
#pragma unroll
    for (int dk = 0; dk < 4; ++dk) {
      v16bf kb0 = *(const v16bf*)&kbuf[cur][lo][dk * 32 + hi * 16];
      v16bf kb1 = *(const v16bf*)&kbuf[cur][16 + lo][dk * 32 + hi * 16];
      s0 = wmma_bf16(qf[dk].v, kb0, s0);
      s1 = wmma_bf16(qf[dk].v, kb1, s1);
    }

    // ---- online softmax: row stats replicated across each 16-lane half ------
    float corr[8];
#pragma unroll
    for (int i = 0; i < 8; ++i) {
      float a = s0[i] * scale;
      float cc = s1[i] * scale;
      float r = fmaxf(a, cc);
      r = fmaxf(r, __shfl_xor(r, 1, 32));
      r = fmaxf(r, __shfl_xor(r, 2, 32));
      r = fmaxf(r, __shfl_xor(r, 4, 32));
      r = fmaxf(r, __shfl_xor(r, 8, 32));
      float mn = fmaxf(mrun[i], r);
      corr[i] = __expf(mrun[i] - mn);
      float p0 = __expf(a - mn);
      float p1 = __expf(cc - mn);
      float rs = p0 + p1;
      rs += __shfl_xor(rs, 1, 32);
      rs += __shfl_xor(rs, 2, 32);
      rs += __shfl_xor(rs, 4, 32);
      rs += __shfl_xor(rs, 8, 32);
      lsum[i] = lsum[i] * corr[i] + rs;
      mrun[i] = mn;
      pbuf[wave][hi * 8 + i][lo]      = f2bf(p0);
      pbuf[wave][hi * 8 + i][16 + lo] = f2bf(p1);
    }

#pragma unroll
    for (int t = 0; t < 8; ++t)
#pragma unroll
      for (int i = 0; i < 8; ++i) accO[t][i] *= corr[i];

    // ---- reload P as A fragment (wave-private LDS; DS ops in-order) ---------
    Frag16 pa;
    pa.h[0] = *(const v8bf*)&pbuf[wave][lo][hi * 16];
    pa.h[1] = *(const v8bf*)&pbuf[wave][lo][hi * 16 + 8];

    // ---- O += P * V  (8 d-tiles of 16, V from LDS) --------------------------
#pragma unroll
    for (int t = 0; t < 8; ++t) {
      v16bf vb = *(const v16bf*)&vbuf[cur][t * 16 + lo][hi * 16];
      accO[t] = wmma_bf16(pa.v, vb, accO[t]);
    }
  }

  // ---- normalize + write OT [b][x][h*128+d] ---------------------------------
  float inv[8];
#pragma unroll
  for (int i = 0; i < 8; ++i) inv[i] = 1.0f / lsum[i];
#pragma unroll
  for (int t = 0; t < 8; ++t)
#pragma unroll
    for (int i = 0; i < 8; ++i) {
      int xg = x0 + hi * 8 + i;
      int d  = t * 16 + lo;
      OT[((size_t)b * NXP + xg) * (NH * OCD) + h * OCD + d] = f2bf(accO[t][i] * inv[i]);
    }
}

// ---------------------------------------------------------------------------
extern "C" void kernel_launch(void* const* d_in, const int* in_sizes, int n_in,
                              void* d_out, int out_size, void* d_ws, size_t ws_size,
                              hipStream_t stream) {
  (void)in_sizes; (void)n_in; (void)out_size; (void)ws_size;
  const float* x  = (const float*)d_in[0];
  const float* y  = (const float*)d_in[1];
  const float* Wq = (const float*)d_in[2];
  const float* Wk = (const float*)d_in[3];
  const float* Wv = (const float*)d_in[4];
  const float* Wo = (const float*)d_in[5];

  char* p = (char*)d_ws;
  auto take = [&](size_t bytes) -> void* {
    void* r = (void*)p;
    p += (bytes + 255) & ~(size_t)255;
    return r;
  };
  __bf16* Wqb = (__bf16*)take((size_t)NH * OCD * QCH * 2);
  __bf16* Wkb = (__bf16*)take((size_t)NH * OCD * KCH * 2);
  __bf16* Wvb = (__bf16*)take((size_t)NH * OCD * KCH * 2);
  __bf16* Wob = (__bf16*)take((size_t)OCD * NH * OCD * 2);
  __bf16* xT  = (__bf16*)take((size_t)BB * NXP * QCH * 2);
  __bf16* yT  = (__bf16*)take((size_t)BB * NYP * KCH * 2);
  __bf16* QT  = (__bf16*)take((size_t)BB * NH * NXP * OCD * 2);
  __bf16* KT  = (__bf16*)take((size_t)BB * NH * NYP * OCD * 2);
  __bf16* Vb  = (__bf16*)take((size_t)BB * NH * OCD * NYP * 2);
  __bf16* OT  = (__bf16*)take((size_t)BB * NXP * NH * OCD * 2);

  auto cdiv = [](int a, int b) { return (a + b - 1) / b; };

  // 1) casts
  cast_bf16_kernel<<<cdiv(NH * OCD * QCH, 256), 256, 0, stream>>>(Wq, Wqb, NH * OCD * QCH);
  cast_bf16_kernel<<<cdiv(NH * OCD * KCH, 256), 256, 0, stream>>>(Wk, Wkb, NH * OCD * KCH);
  cast_bf16_kernel<<<cdiv(NH * OCD * KCH, 256), 256, 0, stream>>>(Wv, Wvb, NH * OCD * KCH);
  cast_bf16_kernel<<<cdiv(OCD * NH * OCD, 256), 256, 0, stream>>>(Wo, Wob, OCD * NH * OCD);
  transpose_cast_kernel<<<dim3(NXP / 32, QCH / 32, BB), dim3(32, 8), 0, stream>>>(x, xT, QCH, NXP);
  transpose_cast_kernel<<<dim3(NYP / 32, KCH / 32, BB), dim3(32, 8), 0, stream>>>(y, yT, KCH, NYP);

  // 2) projections (WMMA): Q^T, K^T, V
  gemm_bf16_wmma<QCH, 0><<<dim3(NXP / 128, (NH * OCD) / 128, BB), 256, 0, stream>>>(
      Wqb, xT, (void*)QT, NH * OCD, NXP);
  gemm_bf16_wmma<KCH, 0><<<dim3(NYP / 128, (NH * OCD) / 128, BB), 256, 0, stream>>>(
      Wkb, yT, (void*)KT, NH * OCD, NYP);
  gemm_bf16_wmma<KCH, 1><<<dim3(NYP / 128, (NH * OCD) / 128, BB), 256, 0, stream>>>(
      Wvb, yT, (void*)Vb, NH * OCD, NYP);

  // 3) flash attention (WMMA + online softmax + async LDS staging)
  attn_flash_kernel<<<dim3(NXP / 128, NH, BB), 256, 0, stream>>>(QT, KT, Vb, OT);

  // 4) output projection (WMMA, f32 out)
  gemm_bf16_wmma<NH * OCD, 2><<<dim3(NXP / 128, 1, BB), 256, 0, stream>>>(
      Wob, OT, d_out, OCD, NXP);
}
